// my_lstm_42296837931810
// MI455X (gfx1250) — matrix operand order repaired and checked by reading, not compiled, not run on hardware
//
#include <hip/hip_runtime.h>
#include <hip/hip_fp16.h>

// ---------------------------------------------------------------------------
// Persistent LSTM for MI455X (gfx1250, wave32, WMMA + async-to-LDS).
//   B=32, T=2048, IN=256, H=256, d_in=512, z-width=1024.
//   4 workgroups x 4 waves. Each WG pins a f16 weight slice [512 x 256 z-cols]
//   (266 KB) in LDS for all 2048 steps. Each wave owns BOTH M tiles so every
//   B fragment is loaded once and used twice (min LDS bytes/WMMA).
//   c-state lives in VGPRs. h exchanged via double-buffered f16 global buffer,
//   staged into LDS with global_load_async_to_lds_b128 (ASYNCcnt), and a
//   per-step atomic grid barrier. GEMM: v_wmma_f32_16x16x32_f16, f32 accum.
// ---------------------------------------------------------------------------

typedef __attribute__((ext_vector_type(16))) _Float16 v16h;
typedef __attribute__((ext_vector_type(8)))  _Float16 v8h;
typedef __attribute__((ext_vector_type(8)))  float    v8f;

#define TLEN    2048
#define BSZ     32
#define HDIM    256
#define DIN     512      // 256 x + 256 h
#define NWG     4        // workgroups (persistent, grid barrier)
#define THREADS 128      // 4 waves of 32
#define WPG     64       // within-gate columns per WG (256/4)
#define LCOLS   256      // local z-columns per WG (4 gates * WPG)
#define KST     520      // padded K stride for weight LDS (halves)
#define AST     520      // padded row stride for activation LDS (halves)
#define SMEM_BYTES ((LCOLS * KST + BSZ * AST) * 2)   // 299,520 B < 320 KB

__device__ __forceinline__ float fsig(float z) {
    return 1.0f / (1.0f + __expf(-z));
}
__device__ __forceinline__ float ftanh(float z) {
    return 2.0f / (1.0f + __expf(-2.0f * z)) - 1.0f;
}

// CDNA5 async DMA: global -> LDS, 16B per lane, tracked by ASYNCcnt.
__device__ __forceinline__ void async_g2lds_b128(unsigned lds_byte_off,
                                                 const void* gptr) {
    asm volatile("global_load_async_to_lds_b128 %0, %1, off"
                 :
                 : "v"(lds_byte_off), "v"((unsigned long long)(uintptr_t)gptr)
                 : "memory");
}
__device__ __forceinline__ void wait_asynccnt0() {
    asm volatile("s_wait_asynccnt 0" ::: "memory");
}

// Zero the barrier counter and both h buffers before each run (graph-safe,
// deterministic across replays).
__global__ void lstm_reset_kernel(unsigned* cnt, _Float16* h0, _Float16* h1) {
    int tid = blockIdx.x * blockDim.x + threadIdx.x;
    if (tid == 0) *cnt = 0u;
    for (int i = tid; i < BSZ * HDIM; i += blockDim.x * gridDim.x) {
        h0[i] = (_Float16)0.0f;
        h1[i] = (_Float16)0.0f;
    }
}

__global__ void __launch_bounds__(THREADS, 1)
lstm_persistent_kernel(const float* __restrict__ x,
                       const float* __restrict__ Wf, const float* __restrict__ bf,
                       const float* __restrict__ Wi, const float* __restrict__ bi,
                       const float* __restrict__ Wo, const float* __restrict__ bo,
                       const float* __restrict__ Wc, const float* __restrict__ bc,
                       float* __restrict__ out,
                       _Float16* __restrict__ h0buf,
                       _Float16* __restrict__ h1buf,
                       unsigned* __restrict__ cnt)
{
    extern __shared__ _Float16 smem[];
    _Float16* Wl = smem;                            // [LCOLS][KST] col-major W slice
    _Float16* Al = smem + (size_t)LCOLS * KST;      // [32][AST] concat(x_t, h) rows

    const int tid  = threadIdx.x;
    const int wg   = blockIdx.x;        // 0..3
    const int lane = tid & 31;
    const int tt   = tid >> 5;          // wave id = within-gate column tile 0..3

    // ---- one-time: stage weight slice into LDS as f16, column-major --------
    for (int g = 0; g < 4; ++g) {
        const float* W = (g == 0) ? Wf : (g == 1) ? Wi : (g == 2) ? Wo : Wc;
        for (int e = tid; e < WPG * DIN; e += THREADS) {
            int c = e >> 9;             // local within-gate column 0..63
            int k = e & 511;            // k row 0..511
            Wl[(size_t)(g * WPG + c) * KST + k] =
                (_Float16)W[(size_t)k * HDIM + wg * WPG + c];
        }
    }

    // Per-lane bias (column depends only on lane%16 for this wave's tile).
    const int wcol = wg * WPG + tt * 16 + (lane & 15);  // h-column 0..255
    const float biasf = bf[wcol];
    const float biasi = bi[wcol];
    const float biaso = bo[wcol];
    const float biasg = bc[wcol];

    // Persistent c-state: one 16x16 f32 tile per M tile (WMMA C/D layout).
    v8f cst[2] = {v8f{}, v8f{}};

    // Fragment base offsets (ISA 7.12.2 layouts).
    const int asel  = (lane >> 4) * 8;          // A: K sub-base 0 or 8
    const int lcb   = tt * 16 + (lane & 15);    // B: lane's column within gate tile
    const int bksel = (lane >> 4) * 8;          // B: K sub-base 0 or 8
    const _Float16* Ab0 = Al + (size_t)(lane & 15) * AST + asel;         // M tile 0
    const _Float16* Ab1 = Al + (size_t)(16 + (lane & 15)) * AST + asel;  // M tile 1

    __syncthreads();   // weights staged

    for (int t = 0; t < TLEN; ++t) {
        const _Float16* hin  = (t & 1) ? h1buf : h0buf;
        _Float16*       hout = (t & 1) ? h0buf : h1buf;

        // ---- 1) kick off async DMA of h_t into LDS (no VGPR round-trip) ----
        for (int e = tid; e < BSZ * 32; e += THREADS) {
            int row = e >> 5;           // batch row 0..31
            int ch  = e & 31;           // 8-half chunk within h
            unsigned loff = (unsigned)((LCOLS * KST + row * AST + 256 + ch * 8) * 2);
            async_g2lds_b128(loff, hin + row * HDIM + ch * 8);
        }

        // ---- 2) convert x_t fp32 -> f16 into LDS while DMA runs ------------
        for (int e = tid; e < BSZ * 32; e += THREADS) {
            int row = e >> 5;
            int ch  = e & 31;
            const float* src = x + (size_t)row * TLEN * 256 + (size_t)t * 256 + ch * 8;
            v8h hx;
            #pragma unroll
            for (int q = 0; q < 8; ++q) hx[q] = (_Float16)src[q];
            *(v8h*)(Al + (size_t)row * AST + ch * 8) = hx;
        }

        wait_asynccnt0();
        __syncthreads();

        // ---- 3) z = A x Wslice : 2 M tiles x 4 gates, K = 512 = 16 x 32 ----
        // B fragments loaded once, used by both M tiles.
        v8f acc[4][2] = {{v8f{}, v8f{}}, {v8f{}, v8f{}},
                         {v8f{}, v8f{}}, {v8f{}, v8f{}}};

        for (int kt = 0; kt < 16; ++kt) {
            union { v16h v; v8h h[2]; } a0, a1;
            a0.h[0] = *(const v8h*)(Ab0 + kt * 32);
            a0.h[1] = *(const v8h*)(Ab0 + kt * 32 + 16);
            a1.h[0] = *(const v8h*)(Ab1 + kt * 32);
            a1.h[1] = *(const v8h*)(Ab1 + kt * 32 + 16);
            #pragma unroll
            for (int g = 0; g < 4; ++g) {
                union { v16h v; v8h h[2]; } b;
                const _Float16* bp =
                    Wl + (size_t)(g * WPG + lcb) * KST + kt * 32 + bksel;
                b.h[0] = *(const v8h*)(bp);
                b.h[1] = *(const v8h*)(bp + 16);
                acc[g][0] = __builtin_amdgcn_wmma_f32_16x16x32_f16(
                    false, a0.v, false, b.v, (short)0, acc[g][0], false, false);
                acc[g][1] = __builtin_amdgcn_wmma_f32_16x16x32_f16(
                    false, a1.v, false, b.v, (short)0, acc[g][1], false, false);
            }
        }

        // ---- 4) gates, c/h update (C/D layout: row = v + 8*(lane/16)) ------
        #pragma unroll
        for (int mm = 0; mm < 2; ++mm) {
            #pragma unroll
            for (int v = 0; v < 8; ++v) {
                int row = mm * 16 + (lane >> 4) * 8 + v;
                float fg = fsig(acc[0][mm][v] + biasf);
                float ig = fsig(acc[1][mm][v] + biasi);
                float og = fsig(acc[2][mm][v] + biaso);
                float gg = ftanh(acc[3][mm][v] + biasg);
                float cv = fg * cst[mm][v] + ig * gg;
                cst[mm][v] = cv;
                float hv = og * ftanh(cv);
                hout[row * HDIM + wcol] = (_Float16)hv;
                // reference emits reversed time order
                out[(size_t)row * (TLEN * HDIM) +
                    (size_t)(TLEN - 1 - t) * HDIM + wcol] = hv;
            }
        }

        // ---- 5) grid barrier (all 4 WGs) -----------------------------------
        __syncthreads();
        if (tid == 0) {
            __threadfence();
            atomicAdd(cnt, 1u);
            const unsigned target = (unsigned)(NWG * (t + 1));
            while (__hip_atomic_load(cnt, __ATOMIC_RELAXED,
                                     __HIP_MEMORY_SCOPE_AGENT) < target) {
                __builtin_amdgcn_s_sleep(1);
            }
            __threadfence();
        }
        __syncthreads();
    }
}

extern "C" void kernel_launch(void* const* d_in, const int* in_sizes, int n_in,
                              void* d_out, int out_size, void* d_ws, size_t ws_size,
                              hipStream_t stream) {
    const float* x  = (const float*)d_in[0];
    const float* Wf = (const float*)d_in[1];
    const float* bf = (const float*)d_in[2];
    const float* Wi = (const float*)d_in[3];
    const float* bi = (const float*)d_in[4];
    const float* Wo = (const float*)d_in[5];
    const float* bo = (const float*)d_in[6];
    const float* Wc = (const float*)d_in[7];
    const float* bc = (const float*)d_in[8];
    float* out = (float*)d_out;

    unsigned char* ws = (unsigned char*)d_ws;
    unsigned* cnt = (unsigned*)ws;                         // barrier counter
    _Float16* h0  = (_Float16*)(ws + 64);                  // 16 KB
    _Float16* h1  = h0 + BSZ * HDIM;                       // 16 KB

    (void)hipFuncSetAttribute((const void*)lstm_persistent_kernel,
                              hipFuncAttributeMaxDynamicSharedMemorySize,
                              SMEM_BYTES);

    lstm_reset_kernel<<<1, 256, 0, stream>>>(cnt, h0, h1);
    lstm_persistent_kernel<<<NWG, THREADS, SMEM_BYTES, stream>>>(
        x, Wf, bf, Wi, bi, Wo, bo, Wc, bc, out, h0, h1, cnt);
}
